// MAMGCN_submodule_24953759990167
// MI455X (gfx1250) — matrix operand order repaired and checked by reading, not compiled, not run on hardware
//
#include <hip/hip_runtime.h>

// ---------------- model constants ----------------
#define B0c 8
#define NNc 170
#define TTc 24
#define SEQc 12
#define PREDc 12
#define Cc 128
#define Hc 8
#define NBc 2
#define FINc 5
#define Ec 256
#define NSc 16
#define DCc 4
#define DRc 8

typedef __attribute__((ext_vector_type(16))) __bf16 v16bf;
typedef __attribute__((ext_vector_type(8)))  float  v8f;

enum { ACT_NONE = 0, ACT_RELU = 1, ACT_SIGMOID = 2, ACT_SOFTPLUS = 3 };

__device__ __forceinline__ unsigned bf16bits(float f) {
  union { float f; unsigned u; } x; x.f = f;
  unsigned u = x.u;
  unsigned r = u + 0x7FFFu + ((u >> 16) & 1u);   // round-to-nearest-even
  return r >> 16;
}
// pack two f32 -> one VGPR holding 2x bf16 (lo in [15:0], hi in [31:16])
__device__ __forceinline__ unsigned pk2bf(float lo, float hi) {
  return bf16bits(lo) | (bf16bits(hi) << 16);
}
union BFRAG { unsigned d[8]; v16bf v; };   // 16 bf16 = 8 dwords

// ============================================================
// WMMA GEMM: C[b] = res + act(A[b](MxK) @ B[b](KxN) + bias)
// one wave32 per 16(M) x 64(N) tile: 4 named accumulators reuse
// one A fragment -> 4x v_wmma_f32_16x16x32_bf16 per k-step of 32.
// Interior tiles: wave-uniform fast path, b128 A loads, packed
// dword bf16 conversion, global_prefetch of next k-slice.
// ============================================================
__global__ void k_gemm_wmma(const float* __restrict__ A, const float* __restrict__ Bm,
                            const float* __restrict__ bias, const float* __restrict__ res,
                            float* __restrict__ C,
                            int M, int N, int K, int lda, int ldc,
                            long long sA, long long sB, long long sC, int act)
{
  const int lane = threadIdx.x & 31;
  const int n0 = blockIdx.x * 64;
  const int m0 = blockIdx.y * 16;
  const int b  = blockIdx.z;
  A  += (long long)b * sA;
  Bm += (long long)b * sB;
  C  += (long long)b * sC;
  if (res) res += (long long)b * sC;

  const int half = lane >> 4;      // 0: lanes 0-15, 1: lanes 16-31
  const int lid  = lane & 15;
  const int ar   = m0 + lid;
  const bool mfull = (m0 + 16 <= M);
  const bool lda4  = ((lda & 3) == 0);

  v8f acc0 = {}, acc1 = {}, acc2 = {}, acc3 = {};

  for (int k0 = 0; k0 < K; k0 += 32) {
    const bool kfull = (k0 + 32 <= K);

    // prefetch next k-slice (speculative, gfx1250 global_prefetch)
    if (k0 + 32 < K) {
      __builtin_prefetch(A + (long long)ar * lda + k0 + 32, 0, 3);
      __builtin_prefetch(Bm + (long long)(k0 + 32 + half * 16) * N + n0 + lid, 0, 3);
    }

    // ---- A fragment (ISA 16-bit A 16x32 layout) ----
    // lanes 0-15 : K = k0+{0..7}, k0+16+{0..7}
    // lanes16-31 : K = k0+8+{0..7}, k0+24+{0..7}
    BFRAG af;
    if (mfull && kfull && lda4) {                 // wave-uniform fast path
      const float* arow = A + (long long)ar * lda + k0 + half * 8;
      float4 v0 = *(const float4*)(arow);
      float4 v1 = *(const float4*)(arow + 4);
      float4 v2 = *(const float4*)(arow + 16);
      float4 v3 = *(const float4*)(arow + 20);
      af.d[0] = pk2bf(v0.x, v0.y);  af.d[1] = pk2bf(v0.z, v0.w);
      af.d[2] = pk2bf(v1.x, v1.y);  af.d[3] = pk2bf(v1.z, v1.w);
      af.d[4] = pk2bf(v2.x, v2.y);  af.d[5] = pk2bf(v2.z, v2.w);
      af.d[6] = pk2bf(v3.x, v3.y);  af.d[7] = pk2bf(v3.z, v3.w);
    } else {                                      // edge tiles only
      #pragma unroll
      for (int j = 0; j < 4; ++j) {
        int ka = k0 + half * 8 + 2 * j;
        int kb = k0 + 16 + half * 8 + 2 * j;
        float a0 = (ar < M && ka < K)     ? A[(long long)ar * lda + ka]     : 0.f;
        float a1 = (ar < M && ka + 1 < K) ? A[(long long)ar * lda + ka + 1] : 0.f;
        float b0 = (ar < M && kb < K)     ? A[(long long)ar * lda + kb]     : 0.f;
        float b1 = (ar < M && kb + 1 < K) ? A[(long long)ar * lda + kb + 1] : 0.f;
        af.d[j]     = pk2bf(a0, a1);
        af.d[4 + j] = pk2bf(b0, b1);
      }
    }

    // ---- 4 B fragments, one wmma each (A reused) ----
    // B layout: col = tile_col + lid; lanes 0-15 rows k0+{0..7,16..23}, lanes 16-31 rows k0+{8..15,24..31}
    const float* bk = Bm + (long long)(k0 + half * 8) * N + n0 + lid;
    #pragma unroll
    for (int tn = 0; tn < 4; ++tn) {
      const int cbase = n0 + tn * 16;
      BFRAG bf;
      if (kfull && (cbase + 16 <= N)) {           // wave-uniform fast path
        const float* bcol = bk + tn * 16;
        #pragma unroll
        for (int j = 0; j < 8; ++j)
          bf.d[j] = pk2bf(bcol[(long long)(2 * j + (j >= 4 ? 8 : 0)) * N], bcol[(long long)(2 * j + 1 + (j >= 4 ? 8 : 0)) * N]);
      } else {
        const int bc = cbase + lid;
        #pragma unroll
        for (int j = 0; j < 8; ++j) {
          int kr = k0 + half * 8 + 2 * j + (j >= 4 ? 8 : 0);
          float f0 = (bc < N && kr < K)     ? Bm[(long long)kr * N + bc]       : 0.f;
          float f1 = (bc < N && kr + 1 < K) ? Bm[(long long)(kr + 1) * N + bc] : 0.f;
          bf.d[j] = pk2bf(f0, f1);
        }
      }
      if      (tn == 0) acc0 = __builtin_amdgcn_wmma_f32_16x16x32_bf16(false, af.v, false, bf.v, (short)0, acc0, false, false);
      else if (tn == 1) acc1 = __builtin_amdgcn_wmma_f32_16x16x32_bf16(false, af.v, false, bf.v, (short)0, acc1, false, false);
      else if (tn == 2) acc2 = __builtin_amdgcn_wmma_f32_16x16x32_bf16(false, af.v, false, bf.v, (short)0, acc2, false, false);
      else              acc3 = __builtin_amdgcn_wmma_f32_16x16x32_bf16(false, af.v, false, bf.v, (short)0, acc3, false, false);
    }
  }

  // ---- epilogue: VGPR i -> row m0 + i + half*8, col (n0 + tn*16) + lid ----
  #pragma unroll
  for (int tn = 0; tn < 4; ++tn) {
    const int col = n0 + tn * 16 + lid;
    if (col < N) {
      float bv = bias ? bias[col] : 0.f;
      #pragma unroll
      for (int i = 0; i < 8; ++i) {
        int row = m0 + i + half * 8;
        if (row < M) {
          float v = (tn == 0 ? acc0[i] : tn == 1 ? acc1[i] : tn == 2 ? acc2[i] : acc3[i]) + bv;
          if (act == ACT_RELU)          v = v > 0.f ? v : 0.f;
          else if (act == ACT_SIGMOID)  v = 1.f / (1.f + __expf(-v));
          else if (act == ACT_SOFTPLUS) v = (v > 20.f) ? v : __logf(1.f + __expf(v));
          if (res) v += res[(long long)row * ldc + col];
          C[(long long)row * ldc + col] = v;
        }
      }
    }
  }
}

// ============================================================
// Fused attention: one wave per (batch, head, query-row).
// Scores in LDS, wave32 shuffle softmax. dh = 16, H = 8.
// ============================================================
__global__ void k_attn(const float* __restrict__ Q, const float* __restrict__ Kb,
                       const float* __restrict__ V, const float* __restrict__ mask,
                       float* __restrict__ O, int Lq, int Lk, float scale)
{
  const int qi = blockIdx.x, h = blockIdx.y, b = blockIdx.z;
  const int lane = threadIdx.x;
  __shared__ float qs[16];
  __shared__ float sc[192];
  const float* qrow = Q + ((long long)b * Lq + qi) * Cc + h * 16;
  if (lane < 16) qs[lane] = qrow[lane];
  __syncthreads();
  float mx = -1e30f;
  for (int k = lane; k < Lk; k += 32) {
    const float* krow = Kb + ((long long)b * Lk + k) * Cc + h * 16;
    float s = 0.f;
    #pragma unroll
    for (int d = 0; d < 16; ++d) s += qs[d] * krow[d];
    s *= scale;
    if (mask && !(mask[(long long)qi * Lk + k] > 0.f)) s = -1e9f;
    sc[k] = s;
    mx = fmaxf(mx, s);
  }
  #pragma unroll
  for (int off = 16; off >= 1; off >>= 1) mx = fmaxf(mx, __shfl_xor(mx, off, 32));
  __syncthreads();
  float sum = 0.f;
  for (int k = lane; k < Lk; k += 32) {
    float e = __expf(sc[k] - mx);
    sc[k] = e;
    sum += e;
  }
  #pragma unroll
  for (int off = 16; off >= 1; off >>= 1) sum += __shfl_xor(sum, off, 32);
  float inv = 1.f / sum;
  __syncthreads();
  if (lane < 16) {
    float acc = 0.f;
    for (int k = 0; k < Lk; ++k)
      acc += sc[k] * V[((long long)b * Lk + k) * Cc + h * 16 + lane];
    O[((long long)b * Lq + qi) * Cc + h * 16 + lane] = acc * inv;
  }
}

// -------- LayerNorm over C=128, one wave per row; optional residual add --------
__global__ void k_layernorm(const float* __restrict__ x, const float* __restrict__ res,
                            const float* __restrict__ g, const float* __restrict__ b,
                            float* __restrict__ y)
{
  const long long r = blockIdx.x;
  const int lane = threadIdx.x;
  float v[4]; float s = 0.f;
  #pragma unroll
  for (int i = 0; i < 4; ++i) {
    long long idx = r * Cc + i * 32 + lane;
    float t = x[idx];
    if (res) t += res[idx];
    v[i] = t; s += t;
  }
  #pragma unroll
  for (int off = 16; off >= 1; off >>= 1) s += __shfl_xor(s, off, 32);
  float mean = s * (1.f / Cc);
  float s2 = 0.f;
  #pragma unroll
  for (int i = 0; i < 4; ++i) { float d = v[i] - mean; s2 += d * d; }
  #pragma unroll
  for (int off = 16; off >= 1; off >>= 1) s2 += __shfl_xor(s2, off, 32);
  float rs = rsqrtf(s2 * (1.f / Cc) + 1e-5f);
  #pragma unroll
  for (int i = 0; i < 4; ++i) {
    int c = i * 32 + lane;
    y[r * Cc + c] = (v[i] - mean) * rs * g[c] + b[c];
  }
}

// -------- RMS norm over C=128 --------
__global__ void k_rms(const float* __restrict__ x, const float* __restrict__ g,
                      float* __restrict__ y)
{
  const long long r = blockIdx.x;
  const int lane = threadIdx.x;
  float v[4]; float s = 0.f;
  #pragma unroll
  for (int i = 0; i < 4; ++i) { float t = x[r * Cc + i * 32 + lane]; v[i] = t; s += t * t; }
  #pragma unroll
  for (int off = 16; off >= 1; off >>= 1) s += __shfl_xor(s, off, 32);
  float rs = rsqrtf(s * (1.f / Cc) + 1e-5f);
  #pragma unroll
  for (int i = 0; i < 4; ++i) {
    int c = i * 32 + lane;
    y[r * Cc + c] = v[i] * rs * g[c];
  }
}

// -------- depthwise causal conv (DC=4) + SiLU; xb = cols [0,E) of (.,512) --------
__global__ void k_dwconv_silu(const float* __restrict__ xz, const float* __restrict__ w,
                              const float* __restrict__ cb, float* __restrict__ out,
                              int T, int ld)
{
  const int s = blockIdx.x;
  const int e = threadIdx.x;     // E = 256 threads
  for (int t = 0; t < T; ++t) {
    float acc = cb[e];
    #pragma unroll
    for (int j = 0; j < DCc; ++j) {
      int tt = t - (DCc - 1) + j;
      if (tt >= 0) acc += xz[((long long)s * T + tt) * ld + e] * w[e * DCc + j];
    }
    float sv = acc / (1.f + __expf(-acc));
    out[((long long)s * T + t) * Ec + e] = sv;
  }
}

// -------- selective scan: one block per sequence, one thread per channel e --------
__global__ void k_mamba_scan(const float* __restrict__ xbc, const float* __restrict__ dbc,
                             const float* __restrict__ delta, const float* __restrict__ zbuf,
                             const float* __restrict__ A_log, const float* __restrict__ Dp,
                             float* __restrict__ y, int T, int ldz, int offz)
{
  const int s = blockIdx.x;
  const int e = threadIdx.x;    // E = 256
  float a[NSc], h[NSc];
  #pragma unroll
  for (int n = 0; n < NSc; ++n) { a[n] = -__expf(A_log[e * NSc + n]); h[n] = 0.f; }
  const float dval = Dp[e];
  for (int t = 0; t < T; ++t) {
    long long r = (long long)s * T + t;
    float dt = delta[r * Ec + e];
    float xv = xbc[r * Ec + e];
    const float* Brow = dbc + r * (DRc + 2 * NSc) + DRc;
    const float* Crow = Brow + NSc;
    float acc = 0.f;
    #pragma unroll
    for (int n = 0; n < NSc; ++n) {
      float da = __expf(dt * a[n]);
      h[n] = da * h[n] + dt * Brow[n] * xv;
      acc += h[n] * Crow[n];
    }
    acc += xv * dval;
    float zv = zbuf[r * ldz + offz + e];
    float sz = zv / (1.f + __expf(-zv));
    y[r * Ec + e] = acc * sz;
  }
}

// -------- misc elementwise kernels --------
__global__ void k_add(const float* a, const float* b, float* o, long long n) {
  long long i = (long long)blockIdx.x * 256 + threadIdx.x;
  if (i < n) o[i] = a[i] + b[i];
}
__global__ void k_gatemix(const float* g, const float* h, const float* x, float* o, long long n) {
  long long i = (long long)blockIdx.x * 256 + threadIdx.x;
  if (i < n) { float gv = g[i]; float v = gv * h[i] + (1.f - gv) * x[i]; o[i] = v > 0.f ? v : 0.f; }
}
__global__ void k_bn(const float* h, const float* m, const float* v, const float* g,
                     const float* b, float* o, long long n) {
  long long i = (long long)blockIdx.x * 256 + threadIdx.x;
  if (i < n) { int c = (int)(i & (Cc - 1)); o[i] = (h[i] - m[c]) * rsqrtf(v[c] + 1e-5f) * g[c] + b[c]; }
}
__global__ void k_add_ste(const float* x, const float* ste, float* o, long long n, int T) {
  long long i = (long long)blockIdx.x * 256 + threadIdx.x;
  if (i < n) { int c = (int)(i & (Cc - 1)); int t = (int)((i >> 7) % T); o[i] = x[i] + ste[t * Cc + c]; }
}
__global__ void k_copy2d(const float* s, float* d, long long rows, int cols,
                         int lds, int ldd, int soff, int doff) {
  long long n = rows * cols;
  long long i = (long long)blockIdx.x * 256 + threadIdx.x;
  if (i < n) { long long r = i / cols; int c = (int)(i % cols);
               d[r * ldd + doff + c] = s[r * lds + soff + c]; }
}
__global__ void k_perm0213(const float* s, float* d, int D1, int D2, long long n) {
  // s: (B,D1,D2,C) -> d: (B,D2,D1,C), enumerated over src
  long long i = (long long)blockIdx.x * 256 + threadIdx.x;
  if (i < n) {
    int c = (int)(i & (Cc - 1));
    long long r = i >> 7;
    int d2 = (int)(r % D2); r /= D2;
    int d1 = (int)(r % D1); long long b = r / D1;
    d[(((b * D2 + d2) * D1) + d1) * Cc + c] = s[i];
  }
}
__global__ void k_flipT(const float* s, float* d, int T, long long n) {
  long long i = (long long)blockIdx.x * 256 + threadIdx.x;
  if (i < n) {
    int c = (int)(i & (Cc - 1));
    long long r = i >> 7;
    int t = (int)(r % T); long long sq = r / T;
    d[i] = s[((sq * T) + (T - 1 - t)) * Cc + c];
  }
}
__global__ void k_slice_time(const float* s, float* d, int Tsrc, int t0, int tn, long long n) {
  long long i = (long long)blockIdx.x * 256 + threadIdx.x;
  if (i < n) {
    int c = (int)(i & (Cc - 1));
    long long r = i >> 7;
    int t = (int)(r % tn); long long bn = r / tn;
    d[i] = s[((bn * Tsrc) + t0 + t) * Cc + c];
  }
}
__global__ void k_slice_ctrl(const float* x, float* o, long long n) {
  long long i = (long long)blockIdx.x * 256 + threadIdx.x;
  if (i < n) {
    int t = (int)(i % SEQc); long long bn = i / SEQc;
    o[i] = x[(bn * TTc + SEQc + t) * FINc + (FINc - 1)];
  }
}
__global__ void k_build_dec(const float* xemb, const float* xctrl, const float* ste,
                            float* o, long long n) {
  long long i = (long long)blockIdx.x * 256 + threadIdx.x;
  if (i < n) {
    int c = (int)(i & (Cc - 1));
    long long r = i >> 7;
    int t = (int)(r % TTc); long long bn = r / TTc;
    float v = (t < SEQc) ? xemb[(bn * TTc + t) * Cc + c]
                         : xctrl[(bn * SEQc + (t - SEQc)) * Cc + c];
    o[i] = v + ste[t * Cc + c];
  }
}
__global__ void k_fd(const float* out2, const float* vf, const float* km,
                     float* q, long long n) {
  long long i = (long long)blockIdx.x * 256 + threadIdx.x;
  if (i < n) {
    float kv = out2[i * 2 + 1];
    int nn = (int)((i / PREDc) % NNc);
    q[i] = kv * vf[nn] * __expf(-kv * km[nn]);
  }
}

// ============================================================
// host side: parameter decoding (jax pytree sorted-key order)
// ============================================================
typedef const float* cfp;
struct PLin  { cfp W, b; };
struct PLN   { cfp b, g; };
struct PAttn { PLin k, o, q, v; };
struct PEncL { PAttn attn; PLin ff1, ff2; PLN ln1, ln2; };
struct PDecL { PAttn ca; PLin ff1, ff2; PLN ln1, ln2, ln3; PAttn sa; };
struct PGcnL { PLin gate, lin; };
struct PMamba{ cfp A_log, D, W_dt, W_in, W_out, W_x, b_dt, b_in, b_out, conv_b, conv_w, rms_g; };
struct PBiM  { PMamba bwd, fwd; PLN ln; PLin proj; };
struct PTffn { cfp bn_b, bn_g, bn_m, bn_v; PLin f1, f2; };
struct PEncB { PEncL gat; PLin gate; PGcnL gcn[2]; PBiM mamba; PTffn sffn; PEncL tat; PTffn tffn; };
struct PDecB { PDecL dec; PGcnL gcn[2]; };
struct PMlp  { PLin l1, l2, l3; };

static cfp   nx(void* const* d, int& i)    { return (cfp)d[i++]; }
static PLin  pLin(void* const* d, int& i)  { PLin l; l.W = nx(d,i); l.b = nx(d,i); return l; }
static PLN   pLN(void* const* d, int& i)   { PLN l; l.b = nx(d,i); l.g = nx(d,i); return l; }
static PAttn pAttn(void* const* d, int& i) { PAttn a; a.k=pLin(d,i); a.o=pLin(d,i); a.q=pLin(d,i); a.v=pLin(d,i); return a; }
static PEncL pEncL(void* const* d, int& i) { PEncL e; e.attn=pAttn(d,i); e.ff1=pLin(d,i); e.ff2=pLin(d,i); e.ln1=pLN(d,i); e.ln2=pLN(d,i); return e; }
static PDecL pDecL(void* const* d, int& i) { PDecL e; e.ca=pAttn(d,i); e.ff1=pLin(d,i); e.ff2=pLin(d,i); e.ln1=pLN(d,i); e.ln2=pLN(d,i); e.ln3=pLN(d,i); e.sa=pAttn(d,i); return e; }
static PGcnL pGcnL(void* const* d, int& i) { PGcnL g; g.gate=pLin(d,i); g.lin=pLin(d,i); return g; }
static PMamba pMamba(void* const* d, int& i) {
  PMamba m; m.A_log=nx(d,i); m.D=nx(d,i); m.W_dt=nx(d,i); m.W_in=nx(d,i); m.W_out=nx(d,i); m.W_x=nx(d,i);
  m.b_dt=nx(d,i); m.b_in=nx(d,i); m.b_out=nx(d,i); m.conv_b=nx(d,i); m.conv_w=nx(d,i); m.rms_g=nx(d,i); return m;
}
static PBiM  pBiM(void* const* d, int& i)  { PBiM b; b.bwd=pMamba(d,i); b.fwd=pMamba(d,i); b.ln=pLN(d,i); b.proj=pLin(d,i); return b; }
static PTffn pTffn(void* const* d, int& i) { PTffn t; t.bn_b=nx(d,i); t.bn_g=nx(d,i); t.bn_m=nx(d,i); t.bn_v=nx(d,i); t.f1=pLin(d,i); t.f2=pLin(d,i); return t; }
static PEncB pEncB(void* const* d, int& i) { PEncB e; e.gat=pEncL(d,i); e.gate=pLin(d,i); e.gcn[0]=pGcnL(d,i); e.gcn[1]=pGcnL(d,i);
                                             e.mamba=pBiM(d,i); e.sffn=pTffn(d,i); e.tat=pEncL(d,i); e.tffn=pTffn(d,i); return e; }
static PDecB pDecB(void* const* d, int& i) { PDecB e; e.dec=pDecL(d,i); e.gcn[0]=pGcnL(d,i); e.gcn[1]=pGcnL(d,i); return e; }
static PMlp  pMlp(void* const* d, int& i)  { PMlp m; m.l1=pLin(d,i); m.l2=pLin(d,i); m.l3=pLin(d,i); return m; }

// ---------------- workspace bump allocator ----------------
struct WS {
  char* base; size_t cap; size_t off; size_t floorv;
  float* alloc(size_t nf) {
    size_t bytes = (nf * sizeof(float) + 255) & ~(size_t)255;
    if (cap && off + bytes > cap) off = floorv;   // wrap transient region
    float* p = (float*)(base + off); off += bytes; return p;
  }
  void mark() { floorv = off; }
};

static inline unsigned gblocks(long long n) { return (unsigned)((n + 255) / 256); }

static void gemm(hipStream_t st, cfp A, cfp B, cfp bias, cfp res, float* C,
                 int M, int N, int K, int lda, int ldc, int act,
                 int batch = 1, long long sA = 0, long long sB = 0, long long sC = 0)
{
  dim3 g((unsigned)((N + 63) / 64), (unsigned)((M + 15) / 16), (unsigned)batch);
  k_gemm_wmma<<<g, dim3(32), 0, st>>>(A, B, bias, res, C, M, N, K, lda, ldc, sA, sB, sC, act);
}

static void mlp3(hipStream_t st, WS& ws, const PMlp& p, cfp in, float* out,
                 int M, int K, int d1, int d2, int dout, int actLast)
{
  float* h1 = ws.alloc((size_t)M * d1);
  gemm(st, in, p.l1.W, p.l1.b, nullptr, h1, M, d1, K, K, d1, ACT_RELU);
  float* h2 = ws.alloc((size_t)M * d2);
  gemm(st, h1, p.l2.W, p.l2.b, nullptr, h2, M, d2, d1, d1, d2, ACT_RELU);
  gemm(st, h2, p.l3.W, p.l3.b, nullptr, out, M, dout, d2, d2, dout, actLast);
}

static void mha(hipStream_t st, WS& ws, const PAttn& p, cfp xq, cfp xkv, cfp mask,
                int nb, int Lq, int Lk, float* out)
{
  const int Mq = nb * Lq, Mk = nb * Lk;
  float* Q = ws.alloc((size_t)Mq * Cc);
  float* K = ws.alloc((size_t)Mk * Cc);
  float* V = ws.alloc((size_t)Mk * Cc);
  gemm(st, xq,  p.q.W, p.q.b, nullptr, Q, Mq, Cc, Cc, Cc, Cc, ACT_NONE);
  gemm(st, xkv, p.k.W, p.k.b, nullptr, K, Mk, Cc, Cc, Cc, Cc, ACT_NONE);
  gemm(st, xkv, p.v.W, p.v.b, nullptr, V, Mk, Cc, Cc, Cc, Cc, ACT_NONE);
  float* Ao = ws.alloc((size_t)Mq * Cc);
  dim3 g((unsigned)Lq, Hc, (unsigned)nb);
  k_attn<<<g, dim3(32), 0, st>>>(Q, K, V, mask, Ao, Lq, Lk, 0.25f);
  gemm(st, Ao, p.o.W, p.o.b, nullptr, out, Mq, Cc, Cc, Cc, Cc, ACT_NONE);
}

static void enc_layer(hipStream_t st, WS& ws, const PEncL& p, float* x, cfp mask, int nb, int L)
{
  const int M = nb * L;
  float* a = ws.alloc((size_t)M * Cc);
  mha(st, ws, p.attn, x, x, mask, nb, L, L, a);
  float* x1 = ws.alloc((size_t)M * Cc);
  k_layernorm<<<(unsigned)M, 32, 0, st>>>(a, x, p.ln1.g, p.ln1.b, x1);
  float* f = ws.alloc((size_t)M * 4 * Cc);
  gemm(st, x1, p.ff1.W, p.ff1.b, nullptr, f, M, 4 * Cc, Cc, Cc, 4 * Cc, ACT_RELU);
  float* f2 = ws.alloc((size_t)M * Cc);
  gemm(st, f, p.ff2.W, p.ff2.b, nullptr, f2, M, Cc, 4 * Cc, 4 * Cc, Cc, ACT_NONE);
  k_layernorm<<<(unsigned)M, 32, 0, st>>>(f2, x1, p.ln2.g, p.ln2.b, x);
}

static void gcn(hipStream_t st, WS& ws, const PGcnL gl[2], float* x, cfp adj, int nb, int N)
{
  const int M = nb * N;
  const long long n = (long long)M * Cc;
  for (int l = 0; l < 2; ++l) {
    float* t = ws.alloc((size_t)M * Cc);
    gemm(st, adj, x, nullptr, nullptr, t, N, Cc, N, N, Cc, ACT_NONE,
         nb, 0, (long long)N * Cc, (long long)N * Cc);
    float* h = ws.alloc((size_t)M * Cc);
    gemm(st, t, gl[l].lin.W, gl[l].lin.b, nullptr, h, M, Cc, Cc, Cc, Cc, ACT_NONE);
    float* g = ws.alloc((size_t)M * Cc);
    gemm(st, x, gl[l].gate.W, gl[l].gate.b, nullptr, g, M, Cc, Cc, Cc, Cc, ACT_SIGMOID);
    k_gatemix<<<gblocks(n), 256, 0, st>>>(g, h, x, x, n);
  }
}

static void mamba_dir(hipStream_t st, WS& ws, const PMamba& p, cfp xin, float* xout, int S, int T)
{
  const int M = S * T;
  float* xn = ws.alloc((size_t)M * Cc);
  k_rms<<<(unsigned)M, 32, 0, st>>>(xin, p.rms_g, xn);
  float* xz = ws.alloc((size_t)M * 2 * Ec);
  gemm(st, xn, p.W_in, p.b_in, nullptr, xz, M, 2 * Ec, Cc, Cc, 2 * Ec, ACT_NONE);
  float* xbc = ws.alloc((size_t)M * Ec);
  k_dwconv_silu<<<(unsigned)S, Ec, 0, st>>>(xz, p.conv_w, p.conv_b, xbc, T, 2 * Ec);
  const int ND = DRc + 2 * NSc;  // 40
  float* dbc = ws.alloc((size_t)M * ND);
  gemm(st, xbc, p.W_x, nullptr, nullptr, dbc, M, ND, Ec, Ec, ND, ACT_NONE);
  float* delta = ws.alloc((size_t)M * Ec);
  gemm(st, dbc, p.W_dt, p.b_dt, nullptr, delta, M, Ec, DRc, ND, Ec, ACT_SOFTPLUS);
  float* y = ws.alloc((size_t)M * Ec);
  k_mamba_scan<<<(unsigned)S, Ec, 0, st>>>(xbc, dbc, delta, xz, p.A_log, p.D, y, T, 2 * Ec, Ec);
  gemm(st, y, p.W_out, p.b_out, xin, xout, M, Cc, Ec, Ec, Cc, ACT_NONE);
}

static void bi_mamba(hipStream_t st, WS& ws, const PBiM& p, cfp x, float* out, int S, int T)
{
  const int M = S * T;
  const long long n = (long long)M * Cc;
  float* x1 = ws.alloc((size_t)M * Cc);
  mamba_dir(st, ws, p.fwd, x, x1, S, T);
  float* xf = ws.alloc((size_t)M * Cc);
  k_flipT<<<gblocks(n), 256, 0, st>>>(x, xf, T, n);
  float* x2f = ws.alloc((size_t)M * Cc);
  mamba_dir(st, ws, p.bwd, xf, x2f, S, T);
  float* x2 = ws.alloc((size_t)M * Cc);
  k_flipT<<<gblocks(n), 256, 0, st>>>(x2f, x2, T, n);
  float* cat = ws.alloc((size_t)M * 2 * Cc);
  k_copy2d<<<gblocks(n), 256, 0, st>>>(x1, cat, M, Cc, Cc, 2 * Cc, 0, 0);
  k_copy2d<<<gblocks(n), 256, 0, st>>>(x2, cat, M, Cc, Cc, 2 * Cc, 0, Cc);
  float* y = ws.alloc((size_t)M * Cc);
  gemm(st, cat, p.proj.W, p.proj.b, nullptr, y, M, Cc, 2 * Cc, 2 * Cc, Cc, ACT_RELU);
  k_layernorm<<<(unsigned)M, 32, 0, st>>>(y, x, p.ln.g, p.ln.b, out);
}

static void tffn(hipStream_t st, WS& ws, const PTffn& p, cfp t1, cfp t2, float* out, int M)
{
  const long long n = (long long)M * Cc;
  float* cat = ws.alloc((size_t)M * 2 * Cc);
  k_copy2d<<<gblocks(n), 256, 0, st>>>(t1, cat, M, Cc, Cc, 2 * Cc, 0, 0);
  k_copy2d<<<gblocks(n), 256, 0, st>>>(t2, cat, M, Cc, Cc, 2 * Cc, 0, Cc);
  float* h = ws.alloc((size_t)M * 2 * Cc);
  gemm(st, cat, p.f1.W, p.f1.b, nullptr, h, M, 2 * Cc, 2 * Cc, 2 * Cc, 2 * Cc, ACT_RELU);
  float* h2 = ws.alloc((size_t)M * Cc);
  gemm(st, h, p.f2.W, p.f2.b, nullptr, h2, M, Cc, 2 * Cc, 2 * Cc, Cc, ACT_NONE);
  k_bn<<<gblocks(n), 256, 0, st>>>(h2, p.bn_m, p.bn_v, p.bn_g, p.bn_b, out, n);
}

static void enc_block(hipStream_t st, WS& ws, const PEncB& p, float* xe,
                      cfp ste, cfp adj, int B, int N, int T)
{
  const int M = B * N * T;
  const long long n = (long long)M * Cc;
  float* xenc = ws.alloc((size_t)M * Cc);
  k_add_ste<<<gblocks(n), 256, 0, st>>>(xe, ste, xenc, n, T);
  float* xm = ws.alloc((size_t)M * Cc);
  bi_mamba(st, ws, p.mamba, xe, xm, B * N, T);
  float* xt = ws.alloc((size_t)M * Cc);
  k_copy2d<<<gblocks(n), 256, 0, st>>>(xenc, xt, M, Cc, Cc, Cc, 0, 0);
  enc_layer(st, ws, p.tat, xt, nullptr, B * N, T);
  float* xencT = ws.alloc((size_t)M * Cc);
  k_perm0213<<<gblocks(n), 256, 0, st>>>(xenc, xencT, N, T, n);
  float* xgT = ws.alloc((size_t)M * Cc);
  k_copy2d<<<gblocks(n), 256, 0, st>>>(xencT, xgT, M, Cc, Cc, Cc, 0, 0);
  gcn(st, ws, p.gcn, xgT, adj, B * T, N);
  float* xg = ws.alloc((size_t)M * Cc);
  k_perm0213<<<gblocks(n), 256, 0, st>>>(xgT, xg, T, N, n);
  float* xsT = ws.alloc((size_t)M * Cc);
  k_copy2d<<<gblocks(n), 256, 0, st>>>(xencT, xsT, M, Cc, Cc, Cc, 0, 0);
  enc_layer(st, ws, p.gat, xsT, adj, B * T, N);
  float* xs = ws.alloc((size_t)M * Cc);
  k_perm0213<<<gblocks(n), 256, 0, st>>>(xsT, xs, T, N, n);
  float* x_s = ws.alloc((size_t)M * Cc);
  tffn(st, ws, p.sffn, xg, xs, x_s, M);
  float* x_t = ws.alloc((size_t)M * Cc);
  tffn(st, ws, p.tffn, xm, xt, x_t, M);
  float* cat = ws.alloc((size_t)M * 2 * Cc);
  k_copy2d<<<gblocks(n), 256, 0, st>>>(x_s, cat, M, Cc, Cc, 2 * Cc, 0, 0);
  k_copy2d<<<gblocks(n), 256, 0, st>>>(x_t, cat, M, Cc, Cc, 2 * Cc, 0, Cc);
  gemm(st, cat, p.gate.W, p.gate.b, xe, xe, M, Cc, 2 * Cc, 2 * Cc, Cc, ACT_RELU);
}

static void dec_layer(hipStream_t st, WS& ws, const PDecL& p, float* x, cfp mem,
                      int nb, int Lq, int Lk)
{
  const int M = nb * Lq;
  float* a = ws.alloc((size_t)M * Cc);
  mha(st, ws, p.sa, x, x, nullptr, nb, Lq, Lq, a);
  float* x1 = ws.alloc((size_t)M * Cc);
  k_layernorm<<<(unsigned)M, 32, 0, st>>>(a, x, p.ln1.g, p.ln1.b, x1);
  float* c = ws.alloc((size_t)M * Cc);
  mha(st, ws, p.ca, x1, mem, nullptr, nb, Lq, Lk, c);
  float* x2 = ws.alloc((size_t)M * Cc);
  k_layernorm<<<(unsigned)M, 32, 0, st>>>(c, x1, p.ln2.g, p.ln2.b, x2);
  float* f = ws.alloc((size_t)M * 4 * Cc);
  gemm(st, x2, p.ff1.W, p.ff1.b, nullptr, f, M, 4 * Cc, Cc, Cc, 4 * Cc, ACT_RELU);
  float* f2 = ws.alloc((size_t)M * Cc);
  gemm(st, f, p.ff2.W, p.ff2.b, nullptr, f2, M, Cc, 4 * Cc, 4 * Cc, Cc, ACT_NONE);
  k_layernorm<<<(unsigned)M, 32, 0, st>>>(f2, x2, p.ln3.g, p.ln3.b, x);
}

static void dec_block(hipStream_t st, WS& ws, const PDecB& p, float* xd, cfp xe,
                      cfp adj, int B, int N, int T)
{
  const int M = B * N * T;
  const long long n = (long long)M * Cc;
  float* xT = ws.alloc((size_t)M * Cc);
  k_perm0213<<<gblocks(n), 256, 0, st>>>(xd, xT, N, T, n);
  gcn(st, ws, p.gcn, xT, adj, B * T, N);
  float* xg = ws.alloc((size_t)M * Cc);
  k_perm0213<<<gblocks(n), 256, 0, st>>>(xT, xg, T, N, n);
  dec_layer(st, ws, p.dec, xg, xe, B * N, T, SEQc);
  k_add<<<gblocks(n), 256, 0, st>>>(xd, xg, xd, n);
}

// ============================================================
extern "C" void kernel_launch(void* const* d_in, const int* in_sizes, int n_in,
                              void* d_out, int out_size, void* d_ws, size_t ws_size,
                              hipStream_t stream)
{
  (void)in_sizes; (void)n_in; (void)out_size;
  const float* x   = (const float*)d_in[0];
  const float* adj = (const float*)d_in[1];
  int i = 2;
  // jax pytree flatten order (sorted dict keys):
  PDecB decb[NBc];  decb[0] = pDecB(d_in, i); decb[1] = pDecB(d_in, i);
  PMlp  emb       = pMlp(d_in, i);
  PEncB encb[NBc]; encb[0] = pEncB(d_in, i); encb[1] = pEncB(d_in, i);
  PMlp  enc_ctrl  = pMlp(d_in, i);
  cfp   fd_km     = nx(d_in, i);
  cfp   fd_vf     = nx(d_in, i);
  PMlp  final_p   = pMlp(d_in, i);
  cfp   ste       = nx(d_in, i);

  WS ws{(char*)d_ws, ws_size, 0, 0};
  const int Mall = B0c * NNc * TTc;      // 32640
  const int Menc = B0c * NNc * SEQc;     // 16320
  float* x_emb = ws.alloc((size_t)Mall * Cc);
  float* x_e   = ws.alloc((size_t)Menc * Cc);
  float* x_d   = ws.alloc((size_t)Mall * Cc);
  ws.mark();

  // embedding MLP: (B,N,24,5) -> (B,N,24,128)
  mlp3(stream, ws, emb, x, x_emb, Mall, FINc, Cc / 2, Cc / 2, Cc, ACT_NONE);

  // encoder input: first SEQ timesteps
  {
    long long n = (long long)Menc * Cc;
    k_slice_time<<<gblocks(n), 256, 0, stream>>>(x_emb, x_e, TTc, 0, SEQc, n);
  }
  for (int bi = 0; bi < NBc; ++bi)
    enc_block(stream, ws, encb[bi], x_e, ste, adj, B0c, NNc, SEQc);

  // control branch
  float* ctrl_in = ws.alloc((size_t)Menc);
  { long long n = Menc; k_slice_ctrl<<<gblocks(n), 256, 0, stream>>>(x, ctrl_in, n); }
  float* x_ctrl = ws.alloc((size_t)Menc * Cc);
  mlp3(stream, ws, enc_ctrl, ctrl_in, x_ctrl, Menc, 1, Cc / 2, Cc / 2, Cc, ACT_NONE);

  // decoder input
  { long long n = (long long)Mall * Cc;
    k_build_dec<<<gblocks(n), 256, 0, stream>>>(x_emb, x_ctrl, ste, x_d, n); }
  for (int bi = 0; bi < NBc; ++bi)
    dec_block(stream, ws, decb[bi], x_d, x_e, adj, B0c, NNc, TTc);

  // final projection on last PRED timesteps -> d_out (out then q)
  float* xlast = ws.alloc((size_t)Menc * Cc);
  { long long n = (long long)Menc * Cc;
    k_slice_time<<<gblocks(n), 256, 0, stream>>>(x_d, xlast, TTc, TTc - PREDc, PREDc, n); }
  float* out2 = (float*)d_out;                       // (B,N,12,2)
  float* qout = (float*)d_out + (size_t)Menc * 2;    // (B,N,12,1)
  mlp3(stream, ws, final_p, xlast, out2, Menc, Cc, Cc / 2, Cc / 2, 2, ACT_RELU);
  { long long n = Menc; k_fd<<<gblocks(n), 256, 0, stream>>>(out2, fd_vf, fd_km, qout, n); }
}
